// Label_Histogram_89163521065189
// MI455X (gfx1250) — matrix operand order repaired and checked by reading, not compiled, not run on hardware
//
#include <hip/hip_runtime.h>
#include <hip/hip_bf16.h>

typedef __attribute__((ext_vector_type(16))) _Float16 v16h;
typedef __attribute__((ext_vector_type(8)))  float    v8f;
typedef __attribute__((ext_vector_type(2)))  float    v2f;

#define N_PTS   200000
#define DD      64          // feature dim
#define HK      64          // H*K cluster centers
#define KCL     16          // K clusters per head
#define HHEADS  4
#define BSEG    2048
#define TAUC    10.0f

#define WAVES_PER_BLOCK 2
#define PTS_PER_TILE    32
#define TILES_PER_WAVE  4
#define PTS_PER_WAVE    (PTS_PER_TILE * TILES_PER_WAVE)   // 128

#define XSTR 65   // padded LDS strides (bank-conflict avoidance, 64 banks)
#define DSTR 65
#define WSTR 17

// ---------------------------------------------------------------- kernel 0
__global__ __launch_bounds__(256) void zero_agg(float* agg, int n) {
  int i = blockIdx.x * blockDim.x + threadIdx.x;
  if (i < n) agg[i] = 0.f;
}

// ---------------------------------------------------------------- kernel 1
// Fused: distance WMMA-GEMM -> student-t soft assignment -> softmax ->
// weighted outer product -> sorted-segment accumulation into agg (global atomics
// only at segment boundaries).
__global__ __launch_bounds__(64) void assign_accum(
    const float* __restrict__ x, const float* __restrict__ kf,
    const int* __restrict__ index, float* __restrict__ agg) {

  __shared__ _Float16 kf16[HK * DD];                        // 8 KB
  __shared__ float knorm2[HK];
  __shared__ float x_lds [WAVES_PER_BLOCK][PTS_PER_TILE * XSTR];
  __shared__ float dot_lds[WAVES_PER_BLOCK][PTS_PER_TILE * DSTR];
  __shared__ float w_lds [WAVES_PER_BLOCK][PTS_PER_TILE * WSTR];

  const int tid  = threadIdx.x;
  const int lane = tid & 31;
  const int wave = tid >> 5;

  // Stage centers as f16 + their squared norms (f32), once per block.
  for (int j = tid; j < HK * DD; j += blockDim.x) kf16[j] = (_Float16)kf[j];
  for (int c = tid; c < HK; c += blockDim.x) {
    float s = 0.f;
    for (int d = 0; d < DD; ++d) { float v = kf[c * DD + d]; s += v * v; }
    knorm2[c] = s;
  }
  __syncthreads();

  // B fragments for dot = x * kf^T  (B is 32x16 = [K-dim][center]).
  // B layout used here (contiguous k, same as A): lane holds column N=lane%16; lanes 0-15 K=0..15,
  // lanes 16-31 K=16..31 (2 halves per VGPR). Loaded once, reused for all tiles.
  v16h bfrag[2][4];
  {
    const int col = lane & 15;
    const int kb  = (lane < 16) ? 0 : 16;
    for (int kc = 0; kc < 2; ++kc)
      for (int nc = 0; nc < 4; ++nc) {
        const _Float16* src = &kf16[(nc * 16 + col) * DD + kc * 32 + kb];
        v16h b;
        #pragma unroll
        for (int i = 0; i < 16; ++i) b[i] = src[i];
        bfrag[kc][nc] = b;
      }
  }

  const long gw    = (long)blockIdx.x * WAVES_PER_BLOCK + wave;
  const long wbase = gw * PTS_PER_WAVE;

  // Per-wave register accumulator: full agg row (16 clusters x 64 dims),
  // lane owns dims {lane, lane+32}.
  float accA[KCL], accB[KCL];
  #pragma unroll
  for (int k = 0; k < KCL; ++k) { accA[k] = 0.f; accB[k] = 0.f; }
  int curseg = -1;

  float* xw = x_lds[wave];
  float* dw = dot_lds[wave];
  float* ww = w_lds[wave];

  const int rloc  = lane & 15;
  const int abase = (lane < 16) ? 0 : 16;  // f16 A-matrix half-wave K base
  const v8f vzero = {0.f,0.f,0.f,0.f,0.f,0.f,0.f,0.f};

  for (int t = 0; t < TILES_PER_WAVE; ++t) {
    const long base = wbase + (long)t * PTS_PER_TILE;

    // ---- (a) stage x tile 32x64 (f32, zero-padded tail) ----
    for (int r = 0; r < PTS_PER_TILE; ++r) {
      const long n = base + r;
      xw[r * XSTR + lane]      = (n < N_PTS) ? x[n * DD + lane]      : 0.f;
      xw[r * XSTR + lane + 32] = (n < N_PTS) ? x[n * DD + lane + 32] : 0.f;
    }
    __syncthreads();

    // ---- (b) WMMA distance dot-products: two 16-row subtiles x 4 center chunks
    for (int st = 0; st < 2; ++st) {
      v8f dacc[4];
      #pragma unroll
      for (int nc = 0; nc < 4; ++nc) dacc[nc] = vzero;
      #pragma unroll
      for (int kc = 0; kc < 2; ++kc) {
        // A frag: row = st*16 + lane%16; halves = K {base..base+15}
        v16h a;
        const float* xr = &xw[(st * 16 + rloc) * XSTR + kc * 32 + abase];
        #pragma unroll
        for (int i = 0; i < 16; ++i) {
          const int off = i;
          a[i] = (_Float16)xr[off];
        }
        #pragma unroll
        for (int nc = 0; nc < 4; ++nc)
          dacc[nc] = __builtin_amdgcn_wmma_f32_16x16x32_f16(
              false, a, false, bfrag[kc][nc], (short)0, dacc[nc], false, false);
      }
      // D layout: VGPR j -> row st*16 + j + (lane<16?0:8), col = lane%16
      const int rowb = st * 16 + ((lane < 16) ? 0 : 8);
      #pragma unroll
      for (int nc = 0; nc < 4; ++nc)
        #pragma unroll
        for (int j = 0; j < 8; ++j)
          dw[(rowb + j) * DSTR + nc * 16 + rloc] = dacc[nc][j];
    }
    __syncthreads();

    // ---- (c) per-point soft assignment (lane <-> point base+lane) ----
    {
      float xn = 0.f;
      const float* xr = &xw[lane * XSTR];
      for (int d = 0; d < DD; ++d) xn += xr[d] * xr[d];
      // per-head sums of t = 1/(1+d2)   (GAMMA=1 -> exponent is exactly -1)
      float Sh[HHEADS];
      for (int h = 0; h < HHEADS; ++h) {
        float s = 0.f;
        for (int k = 0; k < KCL; ++k) {
          const int c = h * KCL + k;
          float d2 = fmaxf(knorm2[c] + xn - 2.f * dw[lane * DSTR + c], 0.f);
          s += 1.f / (1.f + d2);
        }
        Sh[h] = s;
      }
      float m[KCL];
      #pragma unroll
      for (int k = 0; k < KCL; ++k) m[k] = 0.f;
      for (int h = 0; h < HHEADS; ++h) {
        const float rs = 1.f / Sh[h];
        for (int k = 0; k < KCL; ++k) {
          const int c = h * KCL + k;
          float d2 = fmaxf(knorm2[c] + xn - 2.f * dw[lane * DSTR + c], 0.f);
          m[k] += (1.f / (1.f + d2)) * rs;
        }
      }
      const float scale = TAUC / (float)HHEADS;
      float mx = -1e30f;
      #pragma unroll
      for (int k = 0; k < KCL; ++k) { m[k] *= scale; mx = fmaxf(mx, m[k]); }
      float se = 0.f;
      #pragma unroll
      for (int k = 0; k < KCL; ++k) { m[k] = __expf(m[k] - mx); se += m[k]; }
      const float rse = 1.f / se;
      #pragma unroll
      for (int k = 0; k < KCL; ++k) ww[lane * WSTR + k] = m[k] * rse;
    }
    __syncthreads();

    // ---- (d) sorted-segment weighted outer-product accumulation ----
    for (int nn = 0; nn < PTS_PER_TILE; ++nn) {
      const long n = base + nn;
      if (n >= N_PTS) break;                 // wave-uniform; no barrier inside
      const int s = index[n];
      if (s != curseg) {                     // wave-uniform branch
        if (curseg >= 0) {
          float* dst = &agg[(long)curseg * (KCL * DD)];
          #pragma unroll
          for (int k = 0; k < KCL; ++k) {
            atomicAdd(&dst[k * DD + lane],      accA[k]);
            atomicAdd(&dst[k * DD + lane + 32], accB[k]);
            accA[k] = 0.f; accB[k] = 0.f;
          }
        }
        curseg = s;
      }
      const float xv0 = xw[nn * XSTR + lane];
      const float xv1 = xw[nn * XSTR + lane + 32];
      #pragma unroll
      for (int k = 0; k < KCL; ++k) {
        const float wk = ww[nn * WSTR + k];  // LDS broadcast
        accA[k] += wk * xv0;
        accB[k] += wk * xv1;
      }
    }
    __syncthreads();
  }

  if (curseg >= 0) {
    float* dst = &agg[(long)curseg * (KCL * DD)];
    #pragma unroll
    for (int k = 0; k < KCL; ++k) {
      atomicAdd(&dst[k * DD + lane],      accA[k]);
      atomicAdd(&dst[k * DD + lane + 32], accB[k]);
    }
  }
}

// ---------------------------------------------------------------- kernel 2
// out = leaky_relu(agg @ W^T + b) : (2048x1024)x(1024x64), fp32 WMMA.
// One 16x16 output tile per wave: 512 waves total.
__global__ __launch_bounds__(256) void gemm_out(
    const float* __restrict__ agg, const float* __restrict__ W,
    const float* __restrict__ bias, float* __restrict__ out) {
  const int lane = threadIdx.x & 31;
  const int wave = threadIdx.x >> 5;
  const int gw   = blockIdx.x * 8 + wave;   // 0..511
  const int tm   = gw >> 2;                 // 0..127 row tile
  const int tn   = gw & 3;                  // 0..3   col tile
  const int col  = lane & 15;
  const int k0   = (lane < 16) ? 0 : 2;     // f32 A/B half-wave K base
  const int arow = tm * 16 + col;           // A: row = lane%16
  const int bcol = tn * 16 + col;           // B: col = lane%16
  v8f acc = {0.f,0.f,0.f,0.f,0.f,0.f,0.f,0.f};

#if __has_builtin(__builtin_amdgcn_wmma_f32_16x16x4_f32)
  for (int kc = 0; kc < 1024 / 4; ++kc) {
    const float* ap = &agg[(long)arow * 1024 + kc * 4 + k0];
    const float* bp = &W  [(long)bcol * 1024 + kc * 4 + k0];
    v2f a = { ap[0], ap[1] };
    v2f b = { bp[0], bp[1] };
    acc = __builtin_amdgcn_wmma_f32_16x16x4_f32(
        false, a, false, b, (short)0, acc, false, false);
  }
#else
  // VALU fallback (should not be needed on gfx1250)
  #pragma unroll
  for (int j = 0; j < 8; ++j) {
    const int row = tm * 16 + j + ((lane < 16) ? 0 : 8);
    float s = 0.f;
    for (int k = 0; k < 1024; ++k) s += agg[(long)row * 1024 + k] * W[(long)bcol * 1024 + k];
    acc[j] = s;
  }
#endif

  const float bv = bias[bcol];
  #pragma unroll
  for (int j = 0; j < 8; ++j) {
    const int row = tm * 16 + j + ((lane < 16) ? 0 : 8);
    float v = acc[j] + bv;
    v = (v > 0.f) ? v : 0.01f * v;
    out[(long)row * 64 + bcol] = v;
  }
}

// ---------------------------------------------------------------- launch
extern "C" void kernel_launch(void* const* d_in, const int* in_sizes, int n_in,
                              void* d_out, int out_size, void* d_ws, size_t ws_size,
                              hipStream_t stream) {
  const float* x     = (const float*)d_in[0];
  // d_in[1] = ref (unused by reference)
  const float* kf    = (const float*)d_in[2];
  const float* W     = (const float*)d_in[3];
  const float* bias  = (const float*)d_in[4];
  const int*   index = (const int*)d_in[5];
  float* out = (float*)d_out;
  float* agg = (float*)d_ws;                 // 2048*1024 f32 = 8 MB scratch

  const int aggN = BSEG * KCL * DD;
  zero_agg<<<(aggN + 255) / 256, 256, 0, stream>>>(agg, aggN);

  const int totalWaves = (N_PTS + PTS_PER_WAVE - 1) / PTS_PER_WAVE;          // 1563
  const int blocks     = (totalWaves + WAVES_PER_BLOCK - 1) / WAVES_PER_BLOCK; // 782
  assign_accum<<<blocks, WAVES_PER_BLOCK * 32, 0, stream>>>(x, kf, index, agg);

  gemm_out<<<64, 256, 0, stream>>>(agg, W, bias, out);
}
